// Model_56693568307234
// MI455X (gfx1250) — compile-verified, matrix operand order checked by reading, not run on hardware
//
#include <hip/hip_runtime.h>
#include <hip/hip_bf16.h>

typedef __attribute__((ext_vector_type(16))) _Float16 v16h;
typedef __attribute__((ext_vector_type(8)))  _Float16 v8h;
typedef __attribute__((ext_vector_type(8)))  float    v8f;

#define T_STEPS 22550
#define T_PAD   22560   // multiple of 16; rows [T_STEPS, T_PAD) zeroed
#define NBLK 32
#define GPB 16          // gates per block (512 / 32)

// ---------------------------------------------------------------------------
// Dilated transposed-conv upsampler stage (parallel, VALU).
// Stage 3 (to_cond) writes f16 cond[Tpad][128] directly (trim x[:,1:-1]).
// ---------------------------------------------------------------------------
__global__ void upconv(const float* __restrict__ in, const float* __restrict__ w,
                       const float* __restrict__ bias, float* __restrict__ out,
                       int Cin, int Cout, int Lin, int Lout, int K, int sc, int e,
                       int in_stride, int in_off, int to_cond,
                       _Float16* __restrict__ cond, int T) {
  int idx = blockIdx.x * 256 + threadIdx.x;
  if (idx >= Cout * Lout) return;
  int t = idx % Lout, c = idx / Lout;
  float acc = bias[c];
  int k0 = (e - t) % sc; if (k0 < 0) k0 += sc;
  for (int k = k0; k < K; k += sc) {
    int j = t - e + k;
    if (j < 0) continue;
    int p = j / sc;
    if (p >= Lin) break;
    const float* wp = w + (size_t)c * Cin * K + k;
    const float* ip = in + in_off + p;
    float a = 0.f;
    for (int ci = 0; ci < Cin; ++ci) a += wp[(size_t)ci * K] * ip[(size_t)ci * in_stride];
    acc += a;
  }
  acc = fmaxf(acc, 0.f);
  if (to_cond) {
    int tt = t - 1;
    if (tt >= 0 && tt < T) cond[(size_t)tt * 128 + c] = (_Float16)acc;
  } else {
    out[(size_t)c * Lout + t] = acc;
  }
}

// Pack Wx[:,3:131] (f32) -> Wxh[1536][128] (f16, 256B-aligned rows).
__global__ void pack_wxh(const float* __restrict__ Wx, _Float16* __restrict__ Wxh) {
  int idx = blockIdx.x * 256 + threadIdx.x;   // 1536*128 total
  int r = idx >> 7, c = idx & 127;
  Wxh[(size_t)r * 128 + c] = (_Float16)Wx[(size_t)r * 131 + 3 + c];
}

// ---------------------------------------------------------------------------
// WMMA f16 GEMM: xsT[r][t] = bx[r] + sum_c Wx[r][3+c] * cond[t][c]
// (Tpad x 1536 x 128). Fragment fill = 4 x global_load_b128 per k-tile:
// per-lane K footprint of a 16-bit A(16x32)/B(32x16) fragment is two
// contiguous 8-element runs {g*16 + laneHalf*8 ...} (ISA 7.12.2).
// C stored transposed: lane's 8 M-values (consecutive t) = one b128 store.
// ---------------------------------------------------------------------------
__global__ __launch_bounds__(256) void xstatic_wmma(const _Float16* __restrict__ cond,
                                                    const _Float16* __restrict__ Wxh,
                                                    const float* __restrict__ bx,
                                                    _Float16* __restrict__ xsT) {
  int lane = threadIdx.x & 31;
  int wid  = threadIdx.x >> 5;
  int t0 = blockIdx.x * 16;
  int r0 = (blockIdx.y * 8 + wid) * 16;
  int lh = (lane >> 4) & 1;
  int tA = t0 + (lane & 15);          // A fragment row (M)
  int r  = r0 + (lane & 15);          // B fragment col (N)
  const _Float16* arow = cond + (size_t)tA * 128;
  const _Float16* brow = Wxh + (size_t)r * 128;
  v8f acc = {};
  #pragma unroll
  for (int kt = 0; kt < 4; ++kt) {
    int kbase = (kt << 5) + lh * 8;   // lane's first 8-run; second at +16
    v8h a0 = *(const v8h*)(arow + kbase);
    v8h a1 = *(const v8h*)(arow + kbase + 16);
    v8h b0 = *(const v8h*)(brow + kbase);
    v8h b1 = *(const v8h*)(brow + kbase + 16);
    v16h a = __builtin_shufflevector(a0, a1, 0,1,2,3,4,5,6,7,8,9,10,11,12,13,14,15);
    v16h b = __builtin_shufflevector(b0, b1, 0,1,2,3,4,5,6,7,8,9,10,11,12,13,14,15);
    acc = __builtin_amdgcn_wmma_f32_16x16x32_f16(false, a, false, b, (short)0, acc, false, false);
  }
  float bias = bx[r];
  v8h outv;
  #pragma unroll
  for (int v = 0; v < 8; ++v) outv[v] = (_Float16)(acc[v] + bias);
  // C layout: VGPR v -> M = v + lh*8; consecutive M = consecutive t.
  *(v8h*)(xsT + (size_t)r * T_PAD + t0 + lh * 8) = outv;
}

// ---------------------------------------------------------------------------
// State init (re-run each launch so graph replay is deterministic).
// Also zeroes the cond padding rows [T_STEPS, T_PAD).
// ---------------------------------------------------------------------------
__global__ void init_state(float* hbuf, float* h1buf, float* ocbuf, float* ofbuf,
                           unsigned* bar, _Float16* cond) {
  int tid = threadIdx.x;
  for (int i = tid; i < 1024; i += 256) hbuf[i] = 0.f;
  for (int i = tid; i < 512; i += 256) { h1buf[i] = 0.f; ocbuf[i] = 0.f; ofbuf[i] = 0.f; }
  if (tid < 16) bar[tid] = 0u;
  for (int i = tid; i < (T_PAD - T_STEPS) * 128; i += 256)
    cond[(size_t)T_STEPS * 128 + i] = (_Float16)0.0f;
}

// ---------------------------------------------------------------------------
// Device-wide sense barrier (all NBLK blocks resident; spin with s_sleep).
// ---------------------------------------------------------------------------
__device__ inline void gbarrier(unsigned* bar, unsigned nblk) {
  volatile unsigned* vgen = bar + 1;
  __syncthreads();
  if (threadIdx.x == 0) {
    __threadfence();
    unsigned g = *vgen;
    unsigned arrived = atomicAdd(bar, 1u) + 1u;
    if (arrived == nblk) {
      *(volatile unsigned*)bar = 0u;
      __threadfence();
      atomicAdd(bar + 1, 1u);
    } else {
      while (*vgen == g) { __builtin_amdgcn_s_sleep(1); }
    }
    __threadfence();
  }
  __syncthreads();
}

// ---------------------------------------------------------------------------
// Persistent autoregressive decode. 32 blocks x 256 threads; each block owns
// 16 gates (rows i, 512+i, 1024+i of Wh/Wx), 16 rows of O1w/O3w and 16 cols of
// O2w/O4w, all pinned in LDS as f16 (~102 KB / WGP). 4 global barriers / step.
// ---------------------------------------------------------------------------
__global__ __launch_bounds__(256) void decode_kernel(
    const float* __restrict__ Wx, const float* __restrict__ bx,
    const float* __restrict__ Wh, const float* __restrict__ bh,
    const float* __restrict__ O1w, const float* __restrict__ O1b,
    const float* __restrict__ O2w, const float* __restrict__ O2b,
    const float* __restrict__ O3w, const float* __restrict__ O3b,
    const float* __restrict__ O4w, const float* __restrict__ O4b,
    const _Float16* __restrict__ cond, const _Float16* __restrict__ xsT, int use_xs,
    float* hbuf, float* h1buf, float* ocbuf, float* ofbuf, unsigned* bar,
    float* samples, float* logits) {
  const int tid = threadIdx.x;
  const int b   = blockIdx.x;
  const int i0  = b * GPB;

  __shared__ _Float16 WhS[3][GPB][512];   // r/z/n rows of Wh slice
  __shared__ _Float16 WxS[3][GPB][128];   // mel columns 3..130 (fallback path)
  __shared__ _Float16 O1s[GPB][256];
  __shared__ _Float16 O2s[256][GPB];
  __shared__ _Float16 O3s[GPB][256];
  __shared__ _Float16 O4s[256][GPB];
  __shared__ float wxc[3][3][GPB];        // [col 0..2][r/z/n][gate]
  __shared__ float bhs[3][GPB], bxs[3][GPB];
  __shared__ float o1bS[GPB], o3bS[GPB];
  __shared__ float hcur[512], mcur[128];
  __shared__ float h1s[256], h2s[256];
  __shared__ float hg[3][GPB], xg[3][GPB], xsv[3][GPB];
  __shared__ float relu1[GPB], relu2[GPB];
  __shared__ float red[256];
  __shared__ int   redi[256];
  __shared__ int   bcast_i;

  // ---- preload weights (f32 -> f16) into LDS ----
  for (int idx = tid; idx < 3 * GPB * 512; idx += 256) {
    int w = idx / (GPB * 512); int rem = idx - w * GPB * 512;
    int j = rem >> 9; int e = rem & 511;
    WhS[w][j][e] = (_Float16)Wh[(size_t)(w * 512 + i0 + j) * 512 + e];
  }
  for (int idx = tid; idx < 3 * GPB * 128; idx += 256) {
    int w = idx / (GPB * 128); int rem = idx - w * GPB * 128;
    int j = rem >> 7; int c = rem & 127;
    WxS[w][j][c] = (_Float16)Wx[(size_t)(w * 512 + i0 + j) * 131 + 3 + c];
  }
  for (int idx = tid; idx < GPB * 256; idx += 256) {
    int j = idx >> 8, m = idx & 255;
    O1s[j][m] = (_Float16)O1w[(size_t)(i0 + j) * 256 + m];
    O3s[j][m] = (_Float16)O3w[(size_t)(i0 + j) * 256 + m];
    int n = idx >> 4, jj = idx & 15;
    O2s[n][jj] = (_Float16)O2w[(size_t)n * 512 + i0 + jj];
    O4s[n][jj] = (_Float16)O4w[(size_t)n * 512 + i0 + jj];
  }
  if (tid < 3 * GPB) {
    int w = tid / GPB, j = tid % GPB;
    int row = w * 512 + i0 + j;
    wxc[0][w][j] = Wx[(size_t)row * 131 + 0];
    wxc[1][w][j] = Wx[(size_t)row * 131 + 1];
    wxc[2][w][j] = Wx[(size_t)row * 131 + 2];
    bhs[w][j] = bh[row];
    bxs[w][j] = bx[row];
  }
  if (tid < GPB) { o1bS[tid] = O1b[i0 + tid]; o3bS[tid] = O3b[i0 + tid]; }
  __syncthreads();

  float c_val = 0.f, f_val = 0.f;
  const int sub = tid & 15;        // 16 threads per dot-product row
  const int rowslot = tid >> 4;    // 16 row slots
  volatile float* vh1 = h1buf;
  volatile float* voc = ocbuf;
  volatile float* vof = ofbuf;

  for (int t = 0; t < T_STEPS; ++t) {
    const int buf = t & 1;
    float* nxt = hbuf + (buf ^ 1) * 512;
    volatile float* vcur = (volatile float*)(hbuf + buf * 512);
    volatile float* vnxt = (volatile float*)nxt;

    // ---- phase 1: hg = Wh_slice @ h (+ optional mel matvec), xg1, h1 ----
    hcur[tid]       = vcur[tid];
    hcur[tid + 256] = vcur[tid + 256];
    if (!use_xs && tid < 128) mcur[tid] = (float)cond[(size_t)t * 128 + tid];
    __syncthreads();

    for (int it = 0; it < 3; ++it) {
      const _Float16* wr = WhS[it][rowslot];
      float p = 0.f;
      #pragma unroll 8
      for (int e = sub * 32; e < sub * 32 + 32; ++e) p += (float)wr[e] * hcur[e];
      for (int m = 8; m >= 1; m >>= 1) p += __shfl_xor(p, m, 32);
      if (sub == 0) hg[it][rowslot] = p + bhs[it][rowslot];
      if (!use_xs) {
        const _Float16* wxr = WxS[it][rowslot];
        float q = 0.f;
        #pragma unroll 8
        for (int c2 = sub * 8; c2 < sub * 8 + 8; ++c2) q += (float)wxr[c2] * mcur[c2];
        for (int m = 8; m >= 1; m >>= 1) q += __shfl_xor(q, m, 32);
        if (sub == 0) xsv[it][rowslot] = q + bxs[it][rowslot];
      }
    }
    __syncthreads();

    if (tid < 48) {
      int w = tid >> 4, j = tid & 15;
      float base = use_xs ? (float)xsT[(size_t)(w * 512 + i0 + j) * T_PAD + t] : xsv[w][j];
      xg[w][j] = base + c_val * wxc[0][w][j] + f_val * wxc[1][w][j];
    }
    __syncthreads();

    if (i0 < 256 && tid < GPB) {      // h1 only needed for coarse half
      int j = tid;
      float r = 1.f / (1.f + __expf(-(xg[0][j] + hg[0][j])));
      float z = 1.f / (1.f + __expf(-(xg[1][j] + hg[1][j])));
      float n = tanhf(xg[2][j] + r * hg[2][j]);
      h1buf[i0 + j] = (1.f - z) * n + z * hcur[i0 + j];
    }
    __threadfence();
    gbarrier(bar, NBLK);

    // zero last step's logit accumulators (all readers passed barrier above)
    if (tid < 8) {
      ocbuf[(buf ^ 1) * 256 + b * 8 + tid] = 0.f;
      ofbuf[(buf ^ 1) * 256 + b * 8 + tid] = 0.f;
    }

    // ---- phase 2: coarse output stack ----
    h1s[tid] = vh1[tid];
    __syncthreads();
    {
      const _Float16* o1r = O1s[rowslot];
      float p = 0.f;
      #pragma unroll 8
      for (int m2 = sub * 16; m2 < sub * 16 + 16; ++m2) p += (float)o1r[m2] * h1s[m2];
      for (int m = 8; m >= 1; m >>= 1) p += __shfl_xor(p, m, 32);
      if (sub == 0) relu1[rowslot] = fmaxf(p + o1bS[rowslot], 0.f);
    }
    __syncthreads();
    {
      float p = (b == 0) ? O2b[tid] : 0.f;
      #pragma unroll
      for (int jj = 0; jj < GPB; ++jj) p += (float)O2s[tid][jj] * relu1[jj];
      atomicAdd(&ocbuf[buf * 256 + tid], p);
    }
    __threadfence();
    gbarrier(bar, NBLK);

    // ---- phase 3: argmax coarse (redundant per block), h2 ----
    float vv = voc[buf * 256 + tid];
    red[tid] = vv; redi[tid] = tid; __syncthreads();
    for (int s = 128; s > 0; s >>= 1) {
      if (tid < s) {
        float o = red[tid + s]; int oi = redi[tid + s];
        if (o > red[tid] || (o == red[tid] && oi < redi[tid])) { red[tid] = o; redi[tid] = oi; }
      }
      __syncthreads();
    }
    if (tid == 0) bcast_i = redi[0];
    __syncthreads();
    int c_cat = bcast_i;
    float c_new = (float)c_cat / 127.5f - 1.f;
    if (tid < 8) logits[(size_t)t * 512 + b * 8 + tid] = voc[buf * 256 + b * 8 + tid];

    if (tid < 48) {
      int w = tid >> 4, j = tid & 15;
      xg[w][j] += c_new * wxc[2][w][j];   // inp2 differs only in element 2
    }
    __syncthreads();
    if (tid < GPB) {
      int j = tid;
      float r = 1.f / (1.f + __expf(-(xg[0][j] + hg[0][j])));
      float z = 1.f / (1.f + __expf(-(xg[1][j] + hg[1][j])));
      float n = tanhf(xg[2][j] + r * hg[2][j]);
      nxt[i0 + j] = (1.f - z) * n + z * hcur[i0 + j];   // carry h2
    }
    __threadfence();
    gbarrier(bar, NBLK);

    // ---- phase 4: fine output stack ----
    h2s[tid] = vnxt[256 + tid];
    __syncthreads();
    {
      const _Float16* o3r = O3s[rowslot];
      float p = 0.f;
      #pragma unroll 8
      for (int m2 = sub * 16; m2 < sub * 16 + 16; ++m2) p += (float)o3r[m2] * h2s[m2];
      for (int m = 8; m >= 1; m >>= 1) p += __shfl_xor(p, m, 32);
      if (sub == 0) relu2[rowslot] = fmaxf(p + o3bS[rowslot], 0.f);
    }
    __syncthreads();
    {
      float p = (b == 0) ? O4b[tid] : 0.f;
      #pragma unroll
      for (int jj = 0; jj < GPB; ++jj) p += (float)O4s[tid][jj] * relu2[jj];
      atomicAdd(&ofbuf[buf * 256 + tid], p);
    }
    __threadfence();
    gbarrier(bar, NBLK);

    float fv2 = vof[buf * 256 + tid];
    red[tid] = fv2; redi[tid] = tid; __syncthreads();
    for (int s = 128; s > 0; s >>= 1) {
      if (tid < s) {
        float o = red[tid + s]; int oi = redi[tid + s];
        if (o > red[tid] || (o == red[tid] && oi < redi[tid])) { red[tid] = o; redi[tid] = oi; }
      }
      __syncthreads();
    }
    if (tid == 0) bcast_i = redi[0];
    __syncthreads();
    int f_cat = bcast_i;
    float f_new = (float)f_cat / 127.5f - 1.f;
    if (tid < 8) logits[(size_t)t * 512 + 256 + b * 8 + tid] = vof[buf * 256 + b * 8 + tid];
    if (b == 0 && tid == 0)
      samples[t] = ((float)c_cat * 256.f + (float)f_cat) / 32767.5f - 1.f;

    c_val = c_new; f_val = f_new;
  }
}

// ---------------------------------------------------------------------------
extern "C" void kernel_launch(void* const* d_in, const int* in_sizes, int n_in,
                              void* d_out, int out_size, void* d_ws, size_t ws_size,
                              hipStream_t stream) {
  (void)in_sizes; (void)n_in; (void)out_size;
  const float* mels = (const float*)d_in[0];
  const float* w0 = (const float*)d_in[1];  const float* b0 = (const float*)d_in[2];
  const float* w1 = (const float*)d_in[3];  const float* b1 = (const float*)d_in[4];
  const float* w2 = (const float*)d_in[5];  const float* b2 = (const float*)d_in[6];
  const float* Wx = (const float*)d_in[7];  const float* bx = (const float*)d_in[8];
  const float* Wh = (const float*)d_in[9];  const float* bh = (const float*)d_in[10];
  const float* O1w = (const float*)d_in[11]; const float* O1b = (const float*)d_in[12];
  const float* O2w = (const float*)d_in[13]; const float* O2b = (const float*)d_in[14];
  const float* O3w = (const float*)d_in[15]; const float* O3b = (const float*)d_in[16];
  const float* O4w = (const float*)d_in[17]; const float* O4b = (const float*)d_in[18];

  const int T = T_STEPS;
  char* ws = (char*)d_ws;
  size_t off = 0;
  auto take = [&](size_t bytes) { size_t o = off; off += (bytes + 255) & ~(size_t)255; return o; };
  float*    y1    = (float*)(ws + take((size_t)128 * 412 * 4));
  float*    y2    = (float*)(ws + take((size_t)128 * 2052 * 4));
  _Float16* cond  = (_Float16*)(ws + take((size_t)T_PAD * 128 * 2));
  _Float16* Wxh   = (_Float16*)(ws + take((size_t)1536 * 128 * 2));
  float*    hbuf  = (float*)(ws + take(2 * 512 * 4));
  float*    h1buf = (float*)(ws + take(512 * 4));
  float*    ocb   = (float*)(ws + take(2 * 256 * 4));
  float*    ofb   = (float*)(ws + take(2 * 256 * 4));
  unsigned* bar   = (unsigned*)(ws + take(256));
  size_t xs_off   = off;
  size_t xs_bytes = (size_t)1536 * T_PAD * 2;
  int use_xs = (xs_off + xs_bytes <= ws_size) ? 1 : 0;
  _Float16* xsT = use_xs ? (_Float16*)(ws + xs_off) : nullptr;

  float* samples = (float*)d_out;
  float* logits  = samples + T;

  init_state<<<1, 256, 0, stream>>>(hbuf, h1buf, ocb, ofb, bar, cond);

  // upsampler: 80x84 -> 128x412 -> 128x2052 -> cond[22550][128] (f16)
  upconv<<<(128 * 412 + 255) / 256, 256, 0, stream>>>(mels, w0, b0, y1,
      80, 128, 84, 412, 11, 5, 3, 86, 1, 0, nullptr, 0);
  upconv<<<(128 * 2052 + 255) / 256, 256, 0, stream>>>(y1, w1, b1, y2,
      128, 128, 412, 2052, 11, 5, 3, 412, 0, 0, nullptr, 0);
  upconv<<<(128 * 22552 + 255) / 256, 256, 0, stream>>>(y2, w2, b2, nullptr,
      128, 128, 2052, 22552, 23, 11, 6, 2052, 0, 1, cond, T);

  if (use_xs) {
    pack_wxh<<<(1536 * 128) / 256, 256, 0, stream>>>(Wx, Wxh);
    dim3 g(T_PAD / 16, 12);   // 1536 cols / (8 waves * 16)
    xstatic_wmma<<<g, 256, 0, stream>>>(cond, Wxh, bx, xsT);
  }

  decode_kernel<<<NBLK, 256, 0, stream>>>(Wx, bx, Wh, bh, O1w, O1b, O2w, O2b,
      O3w, O3b, O4w, O4b, cond, xsT, use_xs, hbuf, h1buf, ocb, ofb, bar,
      samples, logits);
}
